// JTNNVAE_27934467293758
// MI455X (gfx1250) — matrix operand order repaired and checked by reading, not compiled, not run on hardware
//
#include <hip/hip_runtime.h>
#include <hip/hip_bf16.h>

// ---------------------------------------------------------------------------
// D-MPNN (chemprop-style) pipeline for MI455X (gfx1250, wave32, WMMA).
// bf16 activations + v_wmma_f32_16x16x32_bf16 with f32 accumulation.
// Line-graph segment_sum exploited as a fixed 3-predecessors-per-edge gather.
// Bandwidth-bound: all big streams (g_m, h, t_agg, binput) kept in bf16.
// ---------------------------------------------------------------------------

typedef __bf16 bf16_t;
typedef __attribute__((ext_vector_type(16))) __bf16 v16bf;
typedef __attribute__((ext_vector_type(8)))  float  v8f;
typedef __attribute__((ext_vector_type(8)))  __bf16 bf8;
typedef __attribute__((ext_vector_type(4)))  __bf16 bf4;
typedef __attribute__((ext_vector_type(4)))  float  f4;

#define HD 256

// ------------------------------- utility kernels ---------------------------

__global__ void k_zero(float* __restrict__ p, long n) {
  long i = (long)blockIdx.x * 256 + threadIdx.x;
  if (i < n) p[i] = 0.f;
}

// f32 [R,C] -> bf16 [R,Cpad] (zero padded)
__global__ void k_cvt_pad(const float* __restrict__ in, bf16_t* __restrict__ out,
                          int R, int C, int Cpad) {
  long i = (long)blockIdx.x * 256 + threadIdx.x;
  long total = (long)R * Cpad;
  if (i >= total) return;
  int r = (int)(i / Cpad), c = (int)(i % Cpad);
  out[i] = (c < C) ? (bf16_t)in[(long)r * C + c] : (bf16_t)0.f;
}

// W [K,N] f32 row-major -> Wt [N,Kpad] bf16 (B col-major for WMMA, zero padded)
__global__ void k_cvt_wt(const float* __restrict__ W, bf16_t* __restrict__ Wt,
                         int K, int Kpad, int N) {
  long i = (long)blockIdx.x * 256 + threadIdx.x;
  long total = (long)N * Kpad;
  if (i >= total) return;
  int n = (int)(i / Kpad), k = (int)(i % Kpad);
  Wt[i] = (k < K) ? (bf16_t)W[(long)k * N + n] : (bf16_t)0.f;
}

// t_agg[e] = sum over the 3 line-graph predecessors of t_m, stored bf16.
__global__ void k_gather_tagg(const float* __restrict__ t_m,
                              const int* __restrict__ lg_src,
                              bf16_t* __restrict__ tagg) {
  int e = blockIdx.x;
  int c = threadIdx.x;  // 0..63, each handles 4 columns
  long p0 = lg_src[3 * e + 0];
  long p1 = lg_src[3 * e + 1];
  long p2 = lg_src[3 * e + 2];
  f4 a = *((const f4*)(t_m + p0 * HD) + c);
  f4 b = *((const f4*)(t_m + p1 * HD) + c);
  f4 d = *((const f4*)(t_m + p2 * HD) + c);
  f4 s = a + b + d;
  bf4 o;
#pragma unroll
  for (int i = 0; i < 4; ++i) o[i] = (bf16_t)s[i];
  *((bf4*)(tagg + (long)e * HD) + c) = o;
}

// h[e] = g_m[pred0] + g_m[pred1] + g_m[pred2] + t_agg[e]  (bf16 in/out, f32 sum)
__global__ void k_gather_h(const bf16_t* __restrict__ g_m,
                           const bf16_t* __restrict__ tagg,
                           const int* __restrict__ lg_src,
                           bf16_t* __restrict__ h) {
  int e = blockIdx.x;
  int c = threadIdx.x;  // 0..63
  long p0 = lg_src[3 * e + 0];
  long p1 = lg_src[3 * e + 1];
  long p2 = lg_src[3 * e + 2];
  bf4 a = *((const bf4*)(g_m + p0 * HD) + c);
  bf4 b = *((const bf4*)(g_m + p1 * HD) + c);
  bf4 d = *((const bf4*)(g_m + p2 * HD) + c);
  bf4 t = *((const bf4*)(tagg + (long)e * HD) + c);
  bf4 o;
#pragma unroll
  for (int i = 0; i < 4; ++i) {
    float s = (float)a[i] + (float)b[i] + (float)d[i] + (float)t[i];
    o[i] = (bf16_t)s;
  }
  *((bf4*)(h + (long)e * HD) + c) = o;
}

// ------------------------------- WMMA GEMM ---------------------------------
// C[M,256] = A[M,K]_bf16 @ Bt[256,K]_bf16^T (+resid bf16)(+bias)(relu)
// Block: 256 threads = 8 waves, each wave does a 16(M) x 64(N) tile
// (4 accumulators of v_wmma_f32_16x16x32_bf16). Bt 64-col slice staged in LDS.
// Epilogue has an unguarded fast path for full tiles (tail block only pays
// the per-row exec-mask guards).
template <int KT, bool RESID, bool BIAS, bool RELU, bool OUTF32, bool OUTRAW, bool OUTBF16>
__global__ __launch_bounds__(256)
void k_gemm(const bf16_t* __restrict__ A, int lda,
            const bf16_t* __restrict__ Bt,       // [256, K] row-major (col-major B)
            const bf16_t* __restrict__ resid,    // [M, 256] bf16 (pre-activation residual)
            const float* __restrict__ bias,      // [256]
            float* __restrict__ outf,            // [M, 256] f32   (post-activation)
            bf16_t* __restrict__ outb,           // [M, 256] bf16  (post-activation)
            bf16_t* __restrict__ outraw,         // [M, 256] bf16  (pre-activation)
            int M) {
  constexpr int K = 32 * KT;
  __shared__ alignas(32) bf16_t sB[64 * K];

  const int nbase = blockIdx.y * 64;

  // Cooperative LDS stage of Bt rows [nbase, nbase+64) (contiguous since ldb==K)
  {
    const bf8* src = (const bf8*)(Bt + (long)nbase * K);
    bf8* dst = (bf8*)sB;
    for (int i = threadIdx.x; i < (64 * K) / 8; i += 256) dst[i] = src[i];
  }
  __syncthreads();

  const int lane = threadIdx.x & 31;
  const int wave = threadIdx.x >> 5;
  const int m0 = blockIdx.x * 128 + wave * 16;
  const int lr = lane & 15;   // row (A) / column (B,C) within tile
  const int hi = lane >> 4;   // lane-half selector

  v8f acc[4] = {};

  int mr = m0 + lr;
  if (mr > M - 1) mr = M - 1;  // clamp tail rows for loads; stores are guarded
  const bf16_t* ap = A + (long)mr * lda + hi * 16;

#pragma unroll
  for (int ks = 0; ks < KT; ++ks) {
    bf8 alo = *(const bf8*)(ap);        // K = ks*32 + hi*16     .. +7
    bf8 ahi = *(const bf8*)(ap + 8);    // K = ks*32 + hi*16 + 8 .. +7
    if (ks + 1 < KT) __builtin_prefetch(ap + 32, 0, 3);  // global_prefetch_b8 (near)
    ap += 32;
    v16bf a;
#pragma unroll
    for (int i = 0; i < 8; ++i) { a[i] = alo[i]; a[8 + i] = ahi[i]; }

    const int kb = ks * 32 + hi * 16;   // B: lanes 0-15 -> K[kb..kb+15] of col lr
#pragma unroll
    for (int t = 0; t < 4; ++t) {
      v16bf b = *(const v16bf*)&sB[(t * 16 + lr) * K + kb];
      acc[t] = __builtin_amdgcn_wmma_f32_16x16x32_bf16(
          false, a, false, b, (short)0, acc[t], false, false);
    }
  }

  // Epilogue: D layout = VGPR r -> row m0 + r + hi*8, lane&15 -> column
  if (m0 + 16 <= M) {
    // ---- fast path: full tile, no per-element guards ----
#pragma unroll
    for (int t = 0; t < 4; ++t) {
      const int n = nbase + t * 16 + lr;
      const float bv = BIAS ? bias[n] : 0.f;
#pragma unroll
      for (int r = 0; r < 8; ++r) {
        const long idx = (long)(m0 + r + hi * 8) * HD + n;
        float v0 = acc[t][r];
        if (RESID) v0 += (float)resid[idx];
        if (BIAS)  v0 += bv;
        float v = RELU ? fmaxf(v0, 0.f) : v0;
        if (OUTF32)  outf[idx] = v;
        if (OUTRAW)  outraw[idx] = (bf16_t)v0;
        if (OUTBF16) outb[idx] = (bf16_t)v;
      }
    }
  } else {
    // ---- tail path: guard each row ----
#pragma unroll
    for (int t = 0; t < 4; ++t) {
      const int n = nbase + t * 16 + lr;
      const float bv = BIAS ? bias[n] : 0.f;
#pragma unroll
      for (int r = 0; r < 8; ++r) {
        const int m = m0 + r + hi * 8;
        if (m < M) {
          const long idx = (long)m * HD + n;
          float v0 = acc[t][r];
          if (RESID) v0 += (float)resid[idx];
          if (BIAS)  v0 += bv;
          float v = RELU ? fmaxf(v0, 0.f) : v0;
          if (OUTF32)  outf[idx] = v;
          if (OUTRAW)  outraw[idx] = (bf16_t)v0;
          if (OUTBF16) outb[idx] = (bf16_t)v;
        }
      }
    }
  }
}

// ------------------------- edge->node / node->graph ------------------------

__global__ void k_node_agg(const bf16_t* __restrict__ g_m,
                           const float* __restrict__ t_m,
                           const int* __restrict__ edge_dst,
                           float* __restrict__ nei) {
  int e = blockIdx.x;
  int c = threadIdx.x;  // 0..255
  int d = edge_dst[e];
  float v = (float)g_m[(long)e * HD + c] + t_m[(long)e * HD + c];
  unsafeAtomicAdd(&nei[(long)d * HD + c], v);  // global_atomic_add_f32
}

__global__ void k_build_ainput(const float* __restrict__ x_nodes,
                               const float* __restrict__ nei,
                               bf16_t* __restrict__ ain) {
  int node = blockIdx.x;
  int c = threadIdx.x;  // 0..319
  float v = (c < 35) ? x_nodes[(long)node * 35 + c]
          : (c < 291) ? nei[(long)node * HD + (c - 35)]
          : 0.f;
  ain[(long)node * 320 + c] = (bf16_t)v;
}

__global__ void k_graph_sum(const float* __restrict__ hid,
                            const int* __restrict__ graph_ids,
                            float* __restrict__ gsum,
                            float* __restrict__ gcnt) {
  int node = blockIdx.x;
  int c = threadIdx.x;  // 0..255
  int g = graph_ids[node];
  unsafeAtomicAdd(&gsum[(long)g * HD + c], hid[(long)node * HD + c]);
  if (c == 0) unsafeAtomicAdd(&gcnt[g], 1.f);
}

__global__ void k_graph_mean(const float* __restrict__ gsum,
                             const float* __restrict__ gcnt,
                             float* __restrict__ out) {
  int g = blockIdx.x;
  int c = threadIdx.x;
  out[(long)g * HD + c] = gsum[(long)g * HD + c] / gcnt[g];
}

// ------------------------------- launcher ----------------------------------

extern "C" void kernel_launch(void* const* d_in, const int* in_sizes, int n_in,
                              void* d_out, int out_size, void* d_ws, size_t ws_size,
                              hipStream_t stream) {
  const float* x_nodes  = (const float*)d_in[0];
  const float* x_edges  = (const float*)d_in[1];
  const float* t_m      = (const float*)d_in[2];
  const float* W_i      = (const float*)d_in[3];
  const float* W_h      = (const float*)d_in[4];
  const float* W_o      = (const float*)d_in[5];
  const float* b_o      = (const float*)d_in[6];
  const int*   edge_dst = (const int*)d_in[7];
  const int*   lg_src   = (const int*)d_in[8];
  // d_in[9] (lg_dst) unused: it is sorted with exactly 3 entries per edge.
  const int*   graph_ids = (const int*)d_in[10];

  const int n = in_sizes[0] / 35;
  const int E = in_sizes[1] / 40;
  const int G = out_size / HD;

  // ---- workspace carve (256B aligned) ----
  char* p = (char*)d_ws;
  auto carve = [&](size_t bytes) -> void* {
    void* r = (void*)p;
    p += (bytes + 255) & ~(size_t)255;
    return r;
  };
  bf16_t* xeb    = (bf16_t*)carve((size_t)E * 64 * 2);    // x_edges bf16 padded K=64
  bf16_t* wti    = (bf16_t*)carve((size_t)256 * 64 * 2);  // W_i^T  [256,64]
  bf16_t* wth    = (bf16_t*)carve((size_t)256 * 256 * 2); // W_h^T  [256,256]
  bf16_t* wto    = (bf16_t*)carve((size_t)256 * 320 * 2); // W_o^T  [256,320]
  bf16_t* binput = (bf16_t*)carve((size_t)E * HD * 2);    // pre-relu residual (bf16)
  bf16_t* g_m    = (bf16_t*)carve((size_t)E * HD * 2);
  bf16_t* hbuf   = (bf16_t*)carve((size_t)E * HD * 2);
  bf16_t* tagg   = (bf16_t*)carve((size_t)E * HD * 2);
  float*  nei    = (float*)carve((size_t)n * HD * 4);
  bf16_t* ain    = (bf16_t*)carve((size_t)n * 320 * 2);
  float*  hid    = (float*)carve((size_t)n * HD * 4);
  float*  gsum   = (float*)carve((size_t)G * HD * 4);
  float*  gcnt   = (float*)carve((size_t)G * 4);
  (void)ws_size; (void)n_in;

  const dim3 b256(256);

  // ---- weight / input conversion (tiny) ----
  k_cvt_wt<<<(256 * 64 + 255) / 256, b256, 0, stream>>>(W_i, wti, 40, 64, 256);
  k_cvt_wt<<<(256 * 256 + 255) / 256, b256, 0, stream>>>(W_h, wth, 256, 256, 256);
  k_cvt_wt<<<(256 * 320 + 255) / 256, b256, 0, stream>>>(W_o, wto, 291, 320, 256);
  k_cvt_pad<<<(int)(((long)E * 64 + 255) / 256), b256, 0, stream>>>(x_edges, xeb, E, 40, 64);

  // ---- binput = x_edges @ W_i (raw bf16) ; g_m = relu(binput) (bf16) ----
  dim3 gE((E + 127) / 128, 4);
  k_gemm<2, false, false, true, false, true, true>
      <<<gE, b256, 0, stream>>>(xeb, 64, wti, nullptr, nullptr,
                                nullptr, g_m, binput, E);

  // ---- t_agg (constant across depth) ----
  k_gather_tagg<<<E, 64, 0, stream>>>(t_m, lg_src, tagg);

  // ---- 5 message-passing iterations ----
  for (int it = 0; it < 5; ++it) {
    k_gather_h<<<E, 64, 0, stream>>>(g_m, tagg, lg_src, hbuf);
    k_gemm<8, true, false, true, false, false, true>
        <<<gE, b256, 0, stream>>>(hbuf, 256, wth, binput, nullptr,
                                  nullptr, g_m, nullptr, E);
  }

  // ---- edge -> node aggregation (g_m + t_m) ----
  k_zero<<<(int)(((long)n * HD + 255) / 256), b256, 0, stream>>>(nei, (long)n * HD);
  k_node_agg<<<E, 256, 0, stream>>>(g_m, t_m, edge_dst, nei);

  // ---- output layer: relu(concat(x_nodes, nei) @ W_o + b_o) ----
  k_build_ainput<<<n, 320, 0, stream>>>(x_nodes, nei, ain);
  dim3 gN((n + 127) / 128, 4);
  k_gemm<10, false, true, true, true, false, false>
      <<<gN, b256, 0, stream>>>(ain, 320, wto, nullptr, b_o,
                                hid, nullptr, nullptr, n);

  // ---- per-graph mean ----
  k_zero<<<(int)(((long)G * HD + 255) / 256), b256, 0, stream>>>(gsum, (long)G * HD);
  k_zero<<<(G + 255) / 256, b256, 0, stream>>>(gcnt, G);
  k_graph_sum<<<n, 256, 0, stream>>>(hid, graph_ids, gsum, gcnt);
  k_graph_mean<<<G, 256, 0, stream>>>(gsum, gcnt, (float*)d_out);
}
